// GraphAttentionNetwork_30597347017035
// MI455X (gfx1250) — compile-verified, matrix operand order checked by reading, not run on hardware
//
#include <hip/hip_runtime.h>
#include <hip/hip_bf16.h>

// ---------------------------------------------------------------------------
// GAT forward for MI455X (gfx1250, wave32).
//
// Roofline: edge_features read (268MB, done ONCE -> 3 et planes) ~= WMMA f16
// compute for all GEMMs (~12.4 GFLOP). Everything matrix-shaped goes through
// v_wmma_f32_16x16x32_f16 (dense throughput shape on CDNA5); softmax /
// layernorm / pooling are f32 VALU. Attention h-tiles are staged into LDS via
// GLOBAL_LOAD_ASYNC_TO_LDS_B128 (ASYNCcnt path) instead of VGPR round-trips.
// ---------------------------------------------------------------------------

typedef _Float16 half_t;
typedef __attribute__((ext_vector_type(16))) _Float16 v16h;
typedef __attribute__((ext_vector_type(8)))  float    v8f;

#define Bc 16
#define Nc 512
#define Ec 16

__device__ __forceinline__ v8f wmma16(v16h a, v16h b, v8f c) {
  return __builtin_amdgcn_wmma_f32_16x16x32_f16(false, a, false, b, (short)0, c,
                                                false, false);
}

// Async global -> LDS 128-bit copy (CDNA5 VGLOBAL async path, ASYNCcnt).
// LDS byte offset = low 32 bits of the generic pointer (AS3->flat mapping puts
// the LDS offset in addr[31:0] per the ISA aperture rules).
__device__ __forceinline__ void async_copy_b128(const void* gsrc, void* ldst) {
  unsigned lo = (unsigned)(unsigned long long)ldst;
  asm volatile("global_load_async_to_lds_b128 %0, %1, off"
               :
               : "v"(lo), "v"(gsrc)
               : "memory");
}
__device__ __forceinline__ void async_wait0() {
  asm volatile("s_wait_asynccnt 0x0" ::: "memory");
}

// fragment index helper: 16-bit A/B layout (ISA 7.12.2)
// lane L: row/col = L&15 ; kbase = (L>>4)*8 ; element i -> k = kbase + (i<8?i:i+8)

// ---------------------------------------------------------------------------
// K0: v = A.sum(0) per layer  (layer0/1: (8,80) -> 80 ; layer2: (1,528) -> 528)
__global__ void k_prep_v(const float* __restrict__ A0, const float* __restrict__ A1,
                         const float* __restrict__ A2, float* __restrict__ vbuf) {
  int t = threadIdx.x;
  if (t < 80) {
    float s0 = 0.f, s1 = 0.f;
    for (int h = 0; h < 8; ++h) { s0 += A0[h * 80 + t]; s1 += A1[h * 80 + t]; }
    vbuf[t] = s0; vbuf[80 + t] = s1;
  }
  for (int j = t; j < 528; j += 256) vbuf[160 + j] = A2[j];
}

// ---------------------------------------------------------------------------
// K1: single streaming pass over edge_features (268MB) producing all 3 et planes.
__global__ void k_et(const float* __restrict__ ef, const float* __restrict__ vbuf,
                     float* __restrict__ et) {
  const long TOT = (long)Bc * Nc * Nc;
  long idx = (long)blockIdx.x * blockDim.x + threadIdx.x;
  if (idx >= TOT) return;
  const float4* p = (const float4*)(ef + idx * Ec);
  float4 q0 = p[0], q1 = p[1], q2 = p[2], q3 = p[3];
  float e[16] = {q0.x,q0.y,q0.z,q0.w, q1.x,q1.y,q1.z,q1.w,
                 q2.x,q2.y,q2.z,q2.w, q3.x,q3.y,q3.z,q3.w};
  const float* v0 = vbuf + 64;        // layer0 v_edge
  const float* v1 = vbuf + 80 + 64;   // layer1 v_edge
  const float* v2 = vbuf + 160 + 512; // layer2 v_edge
  float d0 = 0.f, d1 = 0.f, d2 = 0.f;
#pragma unroll
  for (int i = 0; i < 16; ++i) {
    d0 += e[i] * v0[i]; d1 += e[i] * v1[i]; d2 += e[i] * v2[i];
  }
  et[idx] = d0; et[TOT + idx] = d1; et[2 * TOT + idx] = d2;
}

// ---------------------------------------------------------------------------
// K2: h = LN(x) @ W^T + b  via WMMA f16.  512 blocks x 256 thr (8 waves).
// Block = 16 rows of (B*N), full 256 output cols; wave w -> cols [32w,32w+32).
__global__ void k_ln_proj(const float* __restrict__ x, const float* __restrict__ g,
                          const float* __restrict__ bt, const float* __restrict__ W,
                          const float* __restrict__ bias, half_t* __restrict__ hout,
                          int fi) {
  __shared__ __align__(16) half_t xl[16][256];
  __shared__ float red[16][16], red2[16][16], smean[16], srstd[16];
  int t = threadIdx.x;
  int row = t >> 4, j = t & 15;
  long bn0 = (long)blockIdx.x * 16;
  const float* xr = x + (bn0 + row) * fi;
  float s = 0.f, s2 = 0.f;
  for (int c = j; c < fi; c += 16) { float v = xr[c]; s += v; s2 += v * v; }
  red[row][j] = s; red2[row][j] = s2;
  __syncthreads();
  if (j == 0) {
    float a = 0.f, b2 = 0.f;
    for (int k = 0; k < 16; ++k) { a += red[row][k]; b2 += red2[row][k]; }
    float m = a / fi;
    smean[row] = m; srstd[row] = rsqrtf(b2 / fi - m * m + 1e-5f);
  }
  __syncthreads();
  float m = smean[row], rs = srstd[row];
  for (int c = j; c < fi; c += 16)
    xl[row][c] = (half_t)((xr[c] - m) * rs * g[c] + bt[c]);
  __syncthreads();

  int w = t >> 5, lane = t & 31;
  int r = lane & 15, kb = (lane >> 4) * 8, c0 = w * 32;
  v8f acc0 = {}, acc1 = {};
  for (int kt = 0; kt < fi; kt += 32) {
    v16h a, b0, b1;
#pragma unroll
    for (int i = 0; i < 16; ++i) {
      int k = kb + (i < 8 ? i : i + 8);
      a[i]  = xl[r][kt + k];
      b0[i] = (half_t)W[(long)(c0 + r) * fi + kt + k];
      b1[i] = (half_t)W[(long)(c0 + 16 + r) * fi + kt + k];
    }
    acc0 = wmma16(a, b0, acc0);
    acc1 = wmma16(a, b1, acc1);
  }
  int moff = (lane >> 4) * 8, nn = lane & 15;
#pragma unroll
  for (int rr = 0; rr < 8; ++rr) {
    long orow = bn0 + rr + moff;
    hout[orow * 256 + c0 + nn]      = (half_t)(acc0[rr] + bias[c0 + nn]);
    hout[orow * 256 + c0 + 16 + nn] = (half_t)(acc1[rr] + bias[c0 + 16 + nn]);
  }
}

// ---------------------------------------------------------------------------
// K3: row/col attention logits:  row[bn,h] = sum_x h[bn,h*hs+x]*v_row[x]
__global__ void k_rowcol(const half_t* __restrict__ h, const float* __restrict__ vr,
                         const float* __restrict__ vc, float* __restrict__ rowb,
                         float* __restrict__ colb, int heads, int hs) {
  long bn = (long)blockIdx.x * blockDim.x + threadIdx.x;
  if (bn >= (long)Bc * Nc) return;
  const half_t* hr = h + bn * 256;
  for (int hd = 0; hd < heads; ++hd) {
    float sr = 0.f, sc = 0.f;
    for (int x = 0; x < hs; ++x) {
      float v = (float)hr[hd * hs + x];
      sr += v * vr[x]; sc += v * vc[x];
    }
    rowb[bn * heads + hd] = sr; colb[bn * heads + hd] = sc;
  }
}

// ---------------------------------------------------------------------------
// K4: masked leaky-softmax attention + aggregation (flash-style 2 pass, no NxN
// tensor materialized).  Block = (b, 16-row n tile); wave w -> 32 output cols.
// h-tiles staged via GLOBAL_LOAD_ASYNC_TO_LDS_B128.
__global__ void k_attn(const half_t* __restrict__ hbuf, const float* __restrict__ rowb,
                       const float* __restrict__ colb, const float* __restrict__ etb,
                       const int* __restrict__ adj, float* __restrict__ agg, int heads) {
  __shared__ __align__(16) half_t ht[32][256];
  __shared__ float smax[16][8], ssum[16][8];
  int t = threadIdx.x;
  int b = blockIdx.x >> 5;
  int n0 = (blockIdx.x & 31) * 16;
  const float* etB = etb + (long)b * Nc * Nc;
  const int*   adB = adj + (long)b * Nc * Nc;

  // pass 1: per (n, head) max & sum of exp(leaky(masked logits))
  for (int it = t; it < 16 * heads; it += blockDim.x) {
    int n = it / heads, hd = it % heads, gn = n0 + n;
    float rv = rowb[((long)b * Nc + gn) * heads + hd];
    const float* etr = etB + (long)gn * Nc;
    const int*   adr = adB + (long)gn * Nc;
    float mx = -3.0e38f;
    for (int mI = 0; mI < Nc; ++mI)
      if (adr[mI]) {
        float l = rv + colb[((long)b * Nc + mI) * heads + hd] + etr[mI];
        l = l >= 0.f ? l : 0.2f * l;
        mx = fmaxf(mx, l);
      }
    float sm = 0.f;
    for (int mI = 0; mI < Nc; ++mI)
      if (adr[mI]) {
        float l = rv + colb[((long)b * Nc + mI) * heads + hd] + etr[mI];
        l = l >= 0.f ? l : 0.2f * l;
        sm += __expf(l - mx);
      }
    smax[n][hd] = mx;
    ssum[n][hd] = sm > 0.f ? 1.0f / sm : 0.f;   // nan_to_num semantics
  }
  __syncthreads();

  int w = t >> 5, lane = t & 31;
  int hd = (heads == 8) ? w : 0;
  int c0 = w * 32;
  int r = lane & 15, kb = (lane >> 4) * 8;
  v8f acc0 = {}, acc1 = {};
  for (int mt = 0; mt < Nc; mt += 32) {
    { // async-stage h[mt:mt+32, 0:256] f16 tile directly into LDS (ASYNCcnt)
      int mm = t >> 3, cg = (t & 7) * 32;
      const half_t* gsrc = hbuf + ((long)(b * Nc + mt + mm)) * 256 + cg;
      half_t* ldst = &ht[mm][cg];
      async_copy_b128(gsrc,      ldst);
      async_copy_b128(gsrc + 8,  ldst + 8);
      async_copy_b128(gsrc + 16, ldst + 16);
      async_copy_b128(gsrc + 24, ldst + 24);
      async_wait0();
    }
    __syncthreads();
    int gn = n0 + r;
    float rv = rowb[((long)b * Nc + gn) * heads + hd];
    float mx = smax[r][hd], is = ssum[r][hd];
    const float* etr = etB + (long)gn * Nc + mt;
    const int*   adr = adB + (long)gn * Nc + mt;
    v16h a, b0, b1;
#pragma unroll
    for (int i = 0; i < 16; ++i) {
      int k = kb + (i < 8 ? i : i + 8);
      float p = 0.f;
      if (adr[k]) {
        float l = rv + colb[((long)b * Nc + mt + k) * heads + hd] + etr[k];
        l = l >= 0.f ? l : 0.2f * l;
        p = __expf(l - mx) * is;
      }
      a[i]  = (half_t)p;
      b0[i] = ht[k][c0 + r];
      b1[i] = ht[k][c0 + 16 + r];
    }
    acc0 = wmma16(a, b0, acc0);
    acc1 = wmma16(a, b1, acc1);
    __syncthreads();
  }
  int moff = (lane >> 4) * 8, nn = lane & 15;
#pragma unroll
  for (int rr = 0; rr < 8; ++rr) {
    long orow = (long)b * Nc + n0 + rr + moff;
    agg[orow * 256 + c0 + nn]      = acc0[rr];
    agg[orow * 256 + c0 + 16 + nn] = acc1[rr];
  }
}

// ---------------------------------------------------------------------------
// K5: out_proj GEMM + residual (GEMM res for layer0, identity otherwise)
//     + optional LN2 + leaky. Block = 16 rows x 256 cols.
__global__ void k_outproj(const float* __restrict__ agg, const float* __restrict__ W,
                          const float* __restrict__ bias, const float* __restrict__ xin,
                          const float* __restrict__ resW, const float* __restrict__ resB,
                          int fir, const float* __restrict__ g2,
                          const float* __restrict__ b2, int do_ln2,
                          float* __restrict__ xout) {
  __shared__ __align__(16) half_t al[16][256];
  __shared__ __align__(16) half_t rl[16][64];
  __shared__ float ot[16][256];
  __shared__ float red[16][16], red2[16][16], smean[16], srstd[16];
  int t = threadIdx.x;
  long bn0 = (long)blockIdx.x * 16;
  int row = t >> 4, j = t & 15;
  for (int c = j; c < 256; c += 16) al[row][c] = (half_t)agg[(bn0 + row) * 256 + c];
  if (resW)
    for (int c = j; c < fir; c += 16) rl[row][c] = (half_t)xin[(bn0 + row) * fir + c];
  __syncthreads();

  int w = t >> 5, lane = t & 31;
  int r = lane & 15, kb = (lane >> 4) * 8, c0 = w * 32;
  v8f acc0 = {}, acc1 = {};
  for (int kt = 0; kt < 256; kt += 32) {
    v16h a, b0, b1;
#pragma unroll
    for (int i = 0; i < 16; ++i) {
      int k = kb + (i < 8 ? i : i + 8);
      a[i]  = al[r][kt + k];
      b0[i] = (half_t)W[(long)(c0 + r) * 256 + kt + k];
      b1[i] = (half_t)W[(long)(c0 + 16 + r) * 256 + kt + k];
    }
    acc0 = wmma16(a, b0, acc0);
    acc1 = wmma16(a, b1, acc1);
  }
  if (resW) { // layer0: res = x @ resW^T + resB  (K = 64)
    for (int kt = 0; kt < fir; kt += 32) {
      v16h a, b0, b1;
#pragma unroll
      for (int i = 0; i < 16; ++i) {
        int k = kb + (i < 8 ? i : i + 8);
        a[i]  = rl[r][kt + k];
        b0[i] = (half_t)resW[(long)(c0 + r) * fir + kt + k];
        b1[i] = (half_t)resW[(long)(c0 + 16 + r) * fir + kt + k];
      }
      acc0 = wmma16(a, b0, acc0);
      acc1 = wmma16(a, b1, acc1);
    }
  }
  int moff = (lane >> 4) * 8, nn = lane & 15;
#pragma unroll
  for (int rr = 0; rr < 8; ++rr) {
    int mm = rr + moff;
    float r0 = acc0[rr] + bias[c0 + nn];
    float r1 = acc1[rr] + bias[c0 + 16 + nn];
    if (resW) { r0 += resB[c0 + nn]; r1 += resB[c0 + 16 + nn]; }
    else {
      r0 += xin[(bn0 + mm) * 256 + c0 + nn];
      r1 += xin[(bn0 + mm) * 256 + c0 + 16 + nn];
    }
    ot[mm][c0 + nn] = r0; ot[mm][c0 + 16 + nn] = r1;
  }
  __syncthreads();
  if (!do_ln2) {
    for (int c = j; c < 256; c += 16) xout[(bn0 + row) * 256 + c] = ot[row][c];
    return;
  }
  float s = 0.f, s2 = 0.f;
  for (int c = j; c < 256; c += 16) { float v = ot[row][c]; s += v; s2 += v * v; }
  red[row][j] = s; red2[row][j] = s2;
  __syncthreads();
  if (j == 0) {
    float a = 0.f, bb = 0.f;
    for (int k = 0; k < 16; ++k) { a += red[row][k]; bb += red2[row][k]; }
    float m = a / 256.f;
    smean[row] = m; srstd[row] = rsqrtf(bb / 256.f - m * m + 1e-5f);
  }
  __syncthreads();
  float m = smean[row], rs = srstd[row];
  for (int c = j; c < 256; c += 16) {
    float v = (ot[row][c] - m) * rs * g2[c] + b2[c];
    xout[(bn0 + row) * 256 + c] = v >= 0.f ? v : 0.2f * v;
  }
}

// ---------------------------------------------------------------------------
// K6: attention pooling + 2-layer MLP head. One block per batch (tiny).
__global__ void k_pool(const float* __restrict__ x, const float* __restrict__ waw,
                       const float* __restrict__ wab, const float* __restrict__ p1w,
                       const float* __restrict__ p1b, const float* __restrict__ p2w,
                       const float* __restrict__ p2b, float* __restrict__ out) {
  int b = blockIdx.x, t = threadIdx.x;
  __shared__ float sc[512], pooled[256], h1[128], rsum;
  const float* xb = x + (long)b * Nc * 256;
  for (int n = t; n < Nc; n += 256) {
    float s = wab[0];
    for (int f = 0; f < 256; ++f) s += xb[n * 256 + f] * waw[f];
    sc[n] = s;
  }
  __syncthreads();
  if (t == 0) {
    float mx = -3.0e38f;
    for (int n = 0; n < Nc; ++n) mx = fmaxf(mx, sc[n]);
    float sm = 0.f;
    for (int n = 0; n < Nc; ++n) { sc[n] = __expf(sc[n] - mx); sm += sc[n]; }
    rsum = 1.f / sm;
  }
  __syncthreads();
  for (int f = t; f < 256; f += 256) {
    float s = 0.f;
    for (int n = 0; n < Nc; ++n) s += sc[n] * xb[n * 256 + f];
    pooled[f] = s * rsum;
  }
  __syncthreads();
  for (int o = t; o < 128; o += 256) {
    float s = p1b[o];
    for (int f = 0; f < 256; ++f) s += pooled[f] * p1w[o * 256 + f];
    h1[o] = fmaxf(s, 0.f);
  }
  __syncthreads();
  if (t == 0) {
    float s = p2b[0];
    for (int o = 0; o < 128; ++o) s += h1[o] * p2w[o];
    out[b] = s;
  }
}

// ---------------------------------------------------------------------------
extern "C" void kernel_launch(void* const* d_in, const int* in_sizes, int n_in,
                              void* d_out, int out_size, void* d_ws, size_t ws_size,
                              hipStream_t stream) {
  // inputs per jax pytree key order (A < ln1_b < ln1_g < ln2_b < ln2_g <
  // out_proj{b,w} < proj{b,w} < res{b,w}):
  const float* node = (const float*)d_in[0];
  const float* edge = (const float*)d_in[1];
  const int*   adj  = (const int*)d_in[2];
  const float *L0A=(const float*)d_in[3],  *L0l1b=(const float*)d_in[4],
              *L0l1g=(const float*)d_in[5], *L0l2b=(const float*)d_in[6],
              *L0l2g=(const float*)d_in[7], *L0opb=(const float*)d_in[8],
              *L0opw=(const float*)d_in[9], *L0pb =(const float*)d_in[10],
              *L0pw =(const float*)d_in[11],*L0rb =(const float*)d_in[12],
              *L0rw =(const float*)d_in[13];
  const float *L1A=(const float*)d_in[14], *L1l1b=(const float*)d_in[15],
              *L1l1g=(const float*)d_in[16],*L1l2b=(const float*)d_in[17],
              *L1l2g=(const float*)d_in[18],*L1opb=(const float*)d_in[19],
              *L1opw=(const float*)d_in[20],*L1pb =(const float*)d_in[21],
              *L1pw =(const float*)d_in[22];
  const float *L2A=(const float*)d_in[23], *L2l1b=(const float*)d_in[24],
              *L2l1g=(const float*)d_in[25],*L2opb=(const float*)d_in[26],
              *L2opw=(const float*)d_in[27],*L2pb =(const float*)d_in[28],
              *L2pw =(const float*)d_in[29];
  const float *p1b=(const float*)d_in[30], *p1w=(const float*)d_in[31],
              *p2b=(const float*)d_in[32], *p2w=(const float*)d_in[33],
              *wab=(const float*)d_in[34], *waw=(const float*)d_in[35];

  // workspace layout (~80.2 MB)
  char* ws = (char*)d_ws;
  const long TOT = (long)Bc * Nc * Nc;
  float*  etbuf = (float*)(ws);                              // 3*TOT f32
  float*  xA    = (float*)(ws + 50331648);                   // B*N*256 f32
  float*  xB    = (float*)(ws + 50331648 + 8388608);
  float*  agg   = (float*)(ws + 50331648 + 2 * 8388608);
  half_t* hbuf  = (half_t*)(ws + 50331648 + 3 * 8388608);    // B*N*256 f16
  float*  rowb  = (float*)(ws + 50331648 + 3 * 8388608 + 4194304);
  float*  colb  = (float*)(ws + 50331648 + 3 * 8388608 + 4194304 + 262144);
  float*  vbuf  = (float*)(ws + 50331648 + 3 * 8388608 + 4194304 + 2 * 262144);

  dim3 blk(256);
  // param prep + single pass over 268MB edge tensor -> all 3 et planes
  k_prep_v<<<1, blk, 0, stream>>>(L0A, L1A, L2A, vbuf);
  k_et<<<(unsigned)((TOT + 255) / 256), blk, 0, stream>>>(edge, vbuf, etbuf);

  // ---- layer 0 (fi=64, heads=8, hs=32, LN2, GEMM residual) ----
  k_ln_proj<<<512, blk, 0, stream>>>(node, L0l1g, L0l1b, L0pw, L0pb, hbuf, 64);
  k_rowcol<<<32, blk, 0, stream>>>(hbuf, vbuf + 0, vbuf + 32, rowb, colb, 8, 32);
  k_attn<<<512, blk, 0, stream>>>(hbuf, rowb, colb, etbuf, adj, agg, 8);
  k_outproj<<<512, blk, 0, stream>>>(agg, L0opw, L0opb, node, L0rw, L0rb, 64,
                                     L0l2g, L0l2b, 1, xA);
  // ---- layer 1 (fi=256, heads=8, hs=32, LN2, identity residual) ----
  k_ln_proj<<<512, blk, 0, stream>>>(xA, L1l1g, L1l1b, L1pw, L1pb, hbuf, 256);
  k_rowcol<<<32, blk, 0, stream>>>(hbuf, vbuf + 80, vbuf + 112, rowb, colb, 8, 32);
  k_attn<<<512, blk, 0, stream>>>(hbuf, rowb, colb, etbuf + TOT, adj, agg, 8);
  k_outproj<<<512, blk, 0, stream>>>(agg, L1opw, L1opb, xA, nullptr, nullptr, 256,
                                     L1l2g, L1l2b, 1, xB);
  // ---- layer 2 (fi=256, heads=1, hs=256, no LN2, identity residual) ----
  k_ln_proj<<<512, blk, 0, stream>>>(xB, L2l1g, L2l1b, L2pw, L2pb, hbuf, 256);
  k_rowcol<<<32, blk, 0, stream>>>(hbuf, vbuf + 160, vbuf + 416, rowb, colb, 1, 256);
  k_attn<<<512, blk, 0, stream>>>(hbuf, rowb, colb, etbuf + 2 * TOT, adj, agg, 1);
  k_outproj<<<512, blk, 0, stream>>>(agg, L2opw, L2opb, xB, nullptr, nullptr, 256,
                                     nullptr, nullptr, 0, xA);
  // ---- pooling head ----
  k_pool<<<Bc, blk, 0, stream>>>(xA, waw, wab, p1w, p1b, p2w, p2b, (float*)d_out);
}
